// GraphAttention_3530463117992
// MI455X (gfx1250) — matrix operand order repaired and checked by reading, not compiled, not run on hardware
//
#include <hip/hip_runtime.h>
#include <hip/hip_bf16.h>
#include <math.h>

typedef __attribute__((ext_vector_type(16))) __bf16 v16bf;
typedef __attribute__((ext_vector_type(8)))  __bf16 v8bf;
typedef __attribute__((ext_vector_type(8)))  float  v8f;

#define B_   4
#define N_   2048
#define D_   512
#define H_   8
#define HD_  64
#define QKV_COLS (3 * D_)      /* 1536 */
#define ROWS     (B_ * N_)     /* 8192 */
#define NW_      (N_ / 32)     /* 64 adjacency words per row */

__device__ __forceinline__ v16bf cat8(v8bf a, v8bf b) {
  return __builtin_shufflevector(a, b, 0,1,2,3,4,5,6,7,8,9,10,11,12,13,14,15);
}

__device__ __forceinline__ v8f wmma_bf16(v16bf a, v16bf b, v8f c) {
  // D = A(16x32 bf16) * B(32x16 bf16) + C(16x16 f32)
  return __builtin_amdgcn_wmma_f32_16x16x32_bf16(
      /*neg_a=*/false, a, /*neg_b=*/false, b,
      /*c_mod=*/(short)0, c, /*reuse_a=*/false, /*reuse_b=*/false);
}

// ---------------- prep kernels ----------------

__global__ void k_cvt_x(const float* __restrict__ x, __bf16* __restrict__ xb, int n) {
  int i = blockIdx.x * 256 + threadIdx.x;
  if (i < n) xb[i] = (__bf16)x[i];
}

// Build Wt[1536][512] bf16, transposed: Wt[n][k] = W{q,k,v}[k][n%512]
__global__ void k_cvt_w(const float* __restrict__ Wq, const float* __restrict__ Wk,
                        const float* __restrict__ Wv, __bf16* __restrict__ wt) {
  int i = blockIdx.x * 256 + threadIdx.x;
  if (i >= QKV_COLS * D_) return;
  int n = i / D_;
  int k = i - n * D_;
  const float* W = (n < D_) ? Wq : (n < 2 * D_) ? Wk : Wv;
  int nl = (n < D_) ? n : (n < 2 * D_) ? (n - D_) : (n - 2 * D_);
  wt[i] = (__bf16)W[(size_t)k * D_ + nl];
}

// adjacency -> bitmask: adjw[i*64 + w] bit b set iff adj[i][w*32+b] != 0
__global__ void k_adj_bits(const float* __restrict__ adj, unsigned* __restrict__ adjw) {
  int i = blockIdx.x * 256 + threadIdx.x;
  if (i >= N_ * NW_) return;
  int r = i / NW_;
  int w = i - r * NW_;
  const float* row = adj + (size_t)r * N_ + (size_t)w * 32;
  unsigned bits = 0u;
  #pragma unroll
  for (int b = 0; b < 32; ++b)
    if (row[b] != 0.0f) bits |= (1u << b);
  adjw[i] = bits;
}

// ---------------- QKV projection (WMMA GEMM) ----------------
// C[8192 x 1536] = Xbf16[8192 x 512] @ Wt^T  (+bias).  One wave -> 16x64 tile.
// Cols 0..511 -> Q (row-major bf16), 512..1023 -> K (row-major bf16),
// 1024..1535 -> V stored transposed as vt[b][h][d][token].
__global__ void __launch_bounds__(32)
k_qkv_gemm(const __bf16* __restrict__ xb, const __bf16* __restrict__ wt,
           const float* __restrict__ bq, const float* __restrict__ bk,
           const float* __restrict__ bvp,
           __bf16* __restrict__ qb, __bf16* __restrict__ kb, __bf16* __restrict__ vt) {
  const int lane = threadIdx.x;
  const int n16  = lane & 15;
  const int half = lane >> 4;
  const int row0 = blockIdx.x * 16;
  const int c0   = blockIdx.y * 64;

  v8f acc[4] = {};
  const __bf16* arow = xb + (size_t)(row0 + n16) * D_;

  for (int k0 = 0; k0 < D_; k0 += 32) {
    // A fragment (16x32): lane r holds K = k0 + half*16 + [0..15]
    v8bf a1 = *(const v8bf*)(arow + k0 + half * 16);
    v8bf a2 = *(const v8bf*)(arow + k0 + half * 16 + 8);
    v16bf af = cat8(a1, a2);
    #pragma unroll
    for (int t = 0; t < 4; ++t) {
      // B fragment (32x16): lane n holds K = k0 + half*16 + [0..15] of column c0+t*16+n
      const __bf16* bp = wt + (size_t)(c0 + t * 16 + n16) * D_ + k0 + half * 16;
      v16bf bf = *(const v16bf*)bp;
      acc[t] = wmma_bf16(af, bf, acc[t]);
    }
  }

  #pragma unroll
  for (int t = 0; t < 4; ++t) {
    const int g = c0 + t * 16 + n16;
    const float bias = (g < D_) ? bq[g] : (g < 2 * D_) ? bk[g - D_] : bvp[g - 2 * D_];
    #pragma unroll
    for (int v = 0; v < 8; ++v) {
      const int row = row0 + v + half * 8;      // C layout: VGPR v, half -> row v+half*8
      const __bf16 val = (__bf16)(acc[t][v] + bias);
      if (g < D_) {
        qb[(size_t)row * D_ + g] = val;
      } else if (g < 2 * D_) {
        kb[(size_t)row * D_ + (g - D_)] = val;
      } else {
        const int dl = g - 2 * D_;
        const int h = dl >> 6, dd = dl & 63;
        const int bb = row >> 11, tok = row & (N_ - 1);
        vt[((size_t)(bb * H_ + h) * HD_ + dd) * N_ + tok] = val;
      }
    }
  }
}

// ---------------- fused flash-attention over the graph mask ----------------
// One wave per (b, h, 16-query-row block). Online softmax over 32-key blocks.
__global__ void __launch_bounds__(32)
k_attn(const __bf16* __restrict__ qb, const __bf16* __restrict__ kb,
       const __bf16* __restrict__ vt, const unsigned* __restrict__ adjw,
       float* __restrict__ out) {
  __shared__ unsigned adjs[16 * NW_];
  __shared__ __align__(32) __bf16 Ps[16 * 32];

  const int lane = threadIdx.x;
  const int n16  = lane & 15;
  const int half = lane >> 4;
  const int i0 = blockIdx.x * 16;
  const int h  = blockIdx.y;
  const int b  = blockIdx.z;

  // Stage this block's 16 adjacency rows (4 KB) into LDS once.
  for (int idx = lane; idx < 16 * NW_; idx += 32)
    adjs[idx] = adjw[(size_t)(i0 + (idx >> 6)) * NW_ + (idx & (NW_ - 1))];
  __syncthreads();

  // Q block (16x64) as two A fragments, kept in registers for the whole j loop.
  const __bf16* qrow = qb + ((size_t)(b * N_ + i0 + n16)) * D_ + h * HD_;
  v16bf qf[2];
  #pragma unroll
  for (int c = 0; c < 2; ++c) {
    v8bf a1 = *(const v8bf*)(qrow + c * 32 + half * 16);
    v8bf a2 = *(const v8bf*)(qrow + c * 32 + half * 16 + 8);
    qf[c] = cat8(a1, a2);
  }

  float m[8], l[8];
  v8f O[4] = {};
  #pragma unroll
  for (int v = 0; v < 8; ++v) { m[v] = -__builtin_inff(); l[v] = 0.0f; }

  const __bf16* kbase = kb + (size_t)b * N_ * D_ + h * HD_;
  const __bf16* vbase = vt + (size_t)(b * H_ + h) * HD_ * N_;

  for (int j0 = 0; j0 < N_; j0 += 32) {
    // ---- S = Q @ K^T for 16x32 key block (two 16x16 C fragments) ----
    v8f S0 = {}, S1 = {};
    #pragma unroll
    for (int c = 0; c < 2; ++c) {
      // B fragment of K^T: lane n holds d = c*32 + half*16 + [0..15] of key j
      v16bf kf0 = *(const v16bf*)(kbase + (size_t)(j0 + n16) * D_ + c * 32 + half * 16);
      v16bf kf1 = *(const v16bf*)(kbase + (size_t)(j0 + 16 + n16) * D_ + c * 32 + half * 16);
      S0 = wmma_bf16(qf[c], kf0, S0);
      S1 = wmma_bf16(qf[c], kf1, S1);
    }

    // ---- mask + scale + online softmax (rows live per-VGPR per-half) ----
    float p0[8], p1[8], alpha[8];
    #pragma unroll
    for (int v = 0; v < 8; ++v) {
      const unsigned word = adjs[(v + half * 8) * NW_ + (j0 >> 5)];
      float s0 = S0[v] * 0.125f + (((word >> n16) & 1u) ? 0.0f : -1e9f);
      float s1 = S1[v] * 0.125f + (((word >> (16 + n16)) & 1u) ? 0.0f : -1e9f);
      float mx = fmaxf(s0, s1);
      #pragma unroll
      for (int off = 1; off < 16; off <<= 1) mx = fmaxf(mx, __shfl_xor(mx, off, 32));
      const float mnew = fmaxf(m[v], mx);
      alpha[v] = __expf(m[v] - mnew);
      p0[v] = __expf(s0 - mnew);
      p1[v] = __expf(s1 - mnew);
      float rs = p0[v] + p1[v];
      #pragma unroll
      for (int off = 1; off < 16; off <<= 1) rs += __shfl_xor(rs, off, 32);
      l[v] = l[v] * alpha[v] + rs;
      m[v] = mnew;
    }

    // ---- relayout P: C fragment -> row-major LDS -> A fragment ----
    __syncthreads();
    #pragma unroll
    for (int v = 0; v < 8; ++v) {
      Ps[(v + half * 8) * 32 + n16]      = (__bf16)p0[v];
      Ps[(v + half * 8) * 32 + 16 + n16] = (__bf16)p1[v];
    }
    __syncthreads();
    v8bf pa = *(const v8bf*)&Ps[n16 * 32 + half * 16];
    v8bf pb = *(const v8bf*)&Ps[n16 * 32 + half * 16 + 8];
    v16bf pf = cat8(pa, pb);

    // ---- O = O*alpha + P @ V  (V transposed -> contiguous B fragments) ----
    #pragma unroll
    for (int t = 0; t < 4; ++t) {
      #pragma unroll
      for (int v = 0; v < 8; ++v) O[t][v] *= alpha[v];
      v16bf vf = *(const v16bf*)(vbase + (size_t)(t * 16 + n16) * N_ + j0 + half * 16);
      O[t] = wmma_bf16(pf, vf, O[t]);
    }
  }

  // ---- normalize + store fp32 output [B, N, H*HD] ----
  #pragma unroll
  for (int v = 0; v < 8; ++v) {
    const float invl = 1.0f / l[v];
    const size_t row = (size_t)(b * N_ + i0 + v + half * 8);
    #pragma unroll
    for (int t = 0; t < 4; ++t)
      out[row * D_ + h * HD_ + t * 16 + n16] = O[t][v] * invl;
  }
}

// ---------------- host launcher ----------------

extern "C" void kernel_launch(void* const* d_in, const int* in_sizes, int n_in,
                              void* d_out, int out_size, void* d_ws, size_t ws_size,
                              hipStream_t stream) {
  const float* x   = (const float*)d_in[0];
  const float* adj = (const float*)d_in[1];
  const float* Wq  = (const float*)d_in[2];
  const float* bq  = (const float*)d_in[3];
  const float* Wk  = (const float*)d_in[4];
  const float* bk  = (const float*)d_in[5];
  const float* Wv  = (const float*)d_in[6];
  const float* bv  = (const float*)d_in[7];
  float* out = (float*)d_out;

  char* ws = (char*)d_ws;
  size_t off = 0;
  auto carve = [&](size_t bytes) -> char* {
    char* p = ws + off;
    off += (bytes + 255) & ~(size_t)255;
    return p;
  };
  __bf16* xb = (__bf16*)carve((size_t)ROWS * D_ * 2);        // 8 MiB
  __bf16* wt = (__bf16*)carve((size_t)QKV_COLS * D_ * 2);    // 1.5 MiB
  __bf16* qb = (__bf16*)carve((size_t)ROWS * D_ * 2);        // 8 MiB
  __bf16* kb = (__bf16*)carve((size_t)ROWS * D_ * 2);        // 8 MiB
  __bf16* vt = (__bf16*)carve((size_t)B_ * H_ * HD_ * N_ * 2); // 8 MiB
  unsigned* adjw = (unsigned*)carve((size_t)N_ * NW_ * 4);   // 512 KiB
  (void)ws_size; (void)in_sizes; (void)n_in; (void)out_size;

  // prep
  {
    int n = ROWS * D_;
    k_cvt_x<<<(n + 255) / 256, 256, 0, stream>>>(x, xb, n);
  }
  {
    int n = QKV_COLS * D_;
    k_cvt_w<<<(n + 255) / 256, 256, 0, stream>>>(Wq, Wk, Wv, wt);
  }
  {
    int n = N_ * NW_;
    k_adj_bits<<<(n + 255) / 256, 256, 0, stream>>>(adj, adjw);
  }
  // QKV projection: grid (row tiles, col tiles of 64)
  {
    dim3 grid(ROWS / 16, QKV_COLS / 64);
    k_qkv_gemm<<<grid, 32, 0, stream>>>(xb, wt, bq, bk, bv, qb, kb, vt);
  }
  // fused masked attention
  {
    dim3 grid(N_ / 16, H_, B_);
    k_attn<<<grid, 32, 0, stream>>>(qb, kb, vt, adjw, out);
  }
}
